// BinarizeLinear_4200478015737
// MI455X (gfx1250) — matrix operand order repaired and checked by reading, not compiled, not run on hardware
//
#include <hip/hip_runtime.h>

// ---------------------------------------------------------------------------
// BinarizeLinear for MI455X (gfx1250, wave32, WMMA)
// out[B,N] = x[B,K] @ sign(W)[N,K]^T + bias[N]
// B=65536, K=1024, N=1024
// fp32 x -> f16 on the fly, sign(W) -> f16 (exact), f32 accumulate via
// V_WMMA_F32_16X16X32_F16. Double-buffered LDS, 1 barrier per k-step,
// non-temporal output stores.
// ---------------------------------------------------------------------------

typedef _Float16 v4h  __attribute__((ext_vector_type(4)));
typedef _Float16 v8h  __attribute__((ext_vector_type(8)));
typedef _Float16 v16h __attribute__((ext_vector_type(16)));
typedef float    v8f  __attribute__((ext_vector_type(8)));
typedef float    v4f  __attribute__((ext_vector_type(4)));

#define M_TOTAL 65536
#define K_TOTAL 1024
#define N_TOTAL 1024

#define BM 256           // workgroup M tile
#define BN 128           // workgroup N tile
#define BK 64            // K step per LDS stage
#define LDSW (BK + 8)    // padded LDS row stride in halves (144 B)

union F16x16 { v16h v; v8h h[2]; };

// ---------------------------------------------------------------------------
// Pass 1: W (fp32) -> sign(W) as f16 {-1,0,+1}  (exact in f16)
// ---------------------------------------------------------------------------
__global__ __launch_bounds__(256)
void binarize_weight_kernel(const float* __restrict__ w,
                            _Float16* __restrict__ wb, int n) {
  int i = blockIdx.x * blockDim.x + threadIdx.x;
  if (i < n) {
    float v = w[i];
    float s = (v > 0.0f) ? 1.0f : ((v < 0.0f) ? -1.0f : 0.0f);
    wb[i] = (_Float16)s;
  }
}

// ---------------------------------------------------------------------------
// Pass 2: tiled WMMA GEMM, double-buffered LDS.
// 256 threads = 8 waves in a 4(M) x 2(N) grid; each wave owns a 64x64 tile
// = 4x4 WMMA 16x16 tiles (128 f32 accumulator VGPRs).
// ---------------------------------------------------------------------------
__global__ __launch_bounds__(256)
void bin_gemm_kernel(const float* __restrict__ x,
                     const _Float16* __restrict__ wb,
                     const float* __restrict__ bias,
                     float* __restrict__ out) {
  __shared__ _Float16 As[2][BM * LDSW];  // 2 x 36 KB
  __shared__ _Float16 Bs[2][BN * LDSW];  // 2 x 18 KB

  const int tid    = threadIdx.x;
  const int lane   = tid & 31;
  const int wave   = tid >> 5;
  const int wave_m = wave & 3;    // 0..3
  const int wave_n = wave >> 2;   // 0..1
  const int lrow   = lane & 15;   // 0..15
  const int lhalf  = lane >> 4;   // 0..1

  const int m0 = blockIdx.y * BM;
  const int n0 = blockIdx.x * BN;

  v8f acc[4][4];
  const v8f vzero = {0.f, 0.f, 0.f, 0.f, 0.f, 0.f, 0.f, 0.f};
#pragma unroll
  for (int mt = 0; mt < 4; ++mt)
#pragma unroll
    for (int nt = 0; nt < 4; ++nt)
      acc[mt][nt] = vzero;

  // ---- prologue: stage tile 0 into buffer 0 ----
  {
#pragma unroll
    for (int i = 0; i < 16; ++i) {
      int c   = tid + 256 * i;
      int row = c >> 4;
      int col = (c & 15) << 2;
      v4f f = *(const v4f*)(x + (size_t)(m0 + row) * K_TOTAL + col);
      v4h h;
      h.x = (_Float16)f.x;
      h.y = (_Float16)f.y;
      h.z = (_Float16)f.z;
      h.w = (_Float16)f.w;
      *(v4h*)(&As[0][row * LDSW + col]) = h;
    }
#pragma unroll
    for (int i = 0; i < 4; ++i) {
      int c   = tid + 256 * i;
      int row = c >> 3;
      int col = (c & 7) << 3;
      *(v8h*)(&Bs[0][row * LDSW + col]) =
          *(const v8h*)(wb + (size_t)(n0 + row) * K_TOTAL + col);
    }
  }
  __syncthreads();

#pragma unroll 2
  for (int k0 = 0; k0 < K_TOTAL; k0 += BK) {
    const int buf = (k0 / BK) & 1;
    const bool has_next = (k0 + BK) < K_TOTAL;

    // ---- issue next-tile global loads into registers (overlaps compute) ----
    v4f afreg[16];
    v8h breg[4];
    if (has_next) {
#pragma unroll
      for (int i = 0; i < 16; ++i) {
        int c   = tid + 256 * i;
        int row = c >> 4;
        int col = (c & 15) << 2;
        afreg[i] =
            *(const v4f*)(x + (size_t)(m0 + row) * K_TOTAL + (k0 + BK) + col);
      }
#pragma unroll
      for (int i = 0; i < 4; ++i) {
        int c   = tid + 256 * i;
        int row = c >> 3;
        int col = (c & 7) << 3;
        breg[i] =
            *(const v8h*)(wb + (size_t)(n0 + row) * K_TOTAL + (k0 + BK) + col);
      }
    }

    // ---- compute from LDS buffer `buf` ----
    const _Float16* __restrict__ Ab = &As[buf][0];
    const _Float16* __restrict__ Bb = &Bs[buf][0];
#pragma unroll
    for (int kk = 0; kk < BK; kk += 32) {
      F16x16 afrag[4], bfrag[4];
      // A fragment (16x32, MxK): lane(m=lrow, half): halves 0-7 = K[8h..8h+8),
      // halves 8-15 = K[16+8h..16+8h+8)   (ISA 7.12.2 16-bit A layout)
#pragma unroll
      for (int mt = 0; mt < 4; ++mt) {
        const _Float16* ap =
            &Ab[(wave_m * 64 + mt * 16 + lrow) * LDSW + kk + 8 * lhalf];
        afrag[mt].h[0] = *(const v8h*)ap;
        afrag[mt].h[1] = *(const v8h*)(ap + 16);
      }
      // B fragment (32x16, KxN): lane(n=lrow, half): halves 0-7 = K[8h..8h+8),
      // halves 8-15 = K[16+8h..16+8h+8); Bs holds B^T row-major.
#pragma unroll
      for (int nt = 0; nt < 4; ++nt) {
        const _Float16* bp =
            &Bb[(wave_n * 64 + nt * 16 + lrow) * LDSW + kk + 8 * lhalf];
        bfrag[nt].h[0] = *(const v8h*)bp;
        bfrag[nt].h[1] = *(const v8h*)(bp + 16);
      }
#pragma unroll
      for (int mt = 0; mt < 4; ++mt)
#pragma unroll
        for (int nt = 0; nt < 4; ++nt)
          acc[mt][nt] = __builtin_amdgcn_wmma_f32_16x16x32_f16(
              /*neg_a=*/false, afrag[mt].v,
              /*neg_b=*/false, bfrag[nt].v,
              /*c_mod=*/(short)0, acc[mt][nt],
              /*reuse_a=*/false, /*reuse_b=*/false);
    }

    // ---- convert + store staged registers into the other buffer ----
    if (has_next) {
      _Float16* __restrict__ An = &As[buf ^ 1][0];
      _Float16* __restrict__ Bn = &Bs[buf ^ 1][0];
#pragma unroll
      for (int i = 0; i < 16; ++i) {
        int c   = tid + 256 * i;
        int row = c >> 4;
        int col = (c & 15) << 2;
        v4h h;
        h.x = (_Float16)afreg[i].x;
        h.y = (_Float16)afreg[i].y;
        h.z = (_Float16)afreg[i].z;
        h.w = (_Float16)afreg[i].w;
        *(v4h*)(&An[row * LDSW + col]) = h;
      }
#pragma unroll
      for (int i = 0; i < 4; ++i) {
        int c   = tid + 256 * i;
        int row = c >> 3;
        int col = (c & 7) << 3;
        *(v8h*)(&Bn[row * LDSW + col]) = breg[i];
      }
    }
    __syncthreads();  // single barrier per k-step (double-buffered)
  }

  // ---- epilogue: C/D layout (ISA 7.12.2): vgpr j -> m = j + 8*lhalf,
  //      n = lrow; add bias, stream out with non-temporal stores ----
#pragma unroll
  for (int nt = 0; nt < 4; ++nt) {
    const int col  = n0 + wave_n * 64 + nt * 16 + lrow;
    const float bv = bias[col];
#pragma unroll
    for (int mt = 0; mt < 4; ++mt) {
      const int row_base = m0 + wave_m * 64 + mt * 16 + 8 * lhalf;
#pragma unroll
      for (int j = 0; j < 8; ++j) {
        __builtin_nontemporal_store(acc[mt][nt][j] + bv,
                                    &out[(size_t)(row_base + j) * N_TOTAL + col]);
      }
    }
  }
}

// ---------------------------------------------------------------------------
extern "C" void kernel_launch(void* const* d_in, const int* in_sizes, int n_in,
                              void* d_out, int out_size, void* d_ws,
                              size_t ws_size, hipStream_t stream) {
  const float* x    = (const float*)d_in[0];
  const float* w    = (const float*)d_in[1];
  const float* bias = (const float*)d_in[2];
  float* out        = (float*)d_out;
  _Float16* wb      = (_Float16*)d_ws;  // 1024*1024*2 B = 2 MB scratch

  const int wn = N_TOTAL * K_TOTAL;
  binarize_weight_kernel<<<(wn + 255) / 256, 256, 0, stream>>>(w, wb, wn);

  dim3 grid(N_TOTAL / BN, M_TOTAL / BM);  // (8, 256)
  bin_gemm_kernel<<<grid, 256, 0, stream>>>(x, wb, bias, out);
}
